// Net_70111046139942
// MI455X (gfx1250) — compile-verified, matrix operand order checked by reading, not run on hardware
//
#include <hip/hip_runtime.h>
#include <hip/hip_bf16.h>

// ---------------------------------------------------------------------------
// 8-layer GATConv forward for MI455X (gfx1250), bf16 WMMA GEMMs + f32 edge ops
// ---------------------------------------------------------------------------

typedef __attribute__((ext_vector_type(16))) __bf16 v16bf;
typedef __attribute__((ext_vector_type(8)))  float  v8f;

#define NEG_SLOPE 0.2f

// ---------------- utility kernels ----------------

__global__ void k_fill_f32(float* __restrict__ p, float v, long long n) {
    long long i = (long long)blockIdx.x * blockDim.x + threadIdx.x;
    if (i < n) p[i] = v;
}

// h f32 [N, din] (row stride din) -> A bf16 [N, din_pad], zero padded
__global__ void k_conv_h_bf16(const float* __restrict__ h, __bf16* __restrict__ A,
                              int N, int din, int din_pad) {
    long long i = (long long)blockIdx.x * blockDim.x + threadIdx.x;
    long long total = (long long)N * din_pad;
    if (i >= total) return;
    int k = (int)(i % din_pad);
    long long n = i / din_pad;
    float v = (k < din) ? h[n * din + k] : 0.0f;
    A[i] = (__bf16)v;
}

// W f32 [din, Kout] row-major -> Wt bf16 [Kp, din_pad] (transposed, zero padded)
__global__ void k_conv_w_bf16t(const float* __restrict__ W, __bf16* __restrict__ Wt,
                               int din, int Kout, int din_pad, int Kp) {
    long long i = (long long)blockIdx.x * blockDim.x + threadIdx.x;
    long long total = (long long)Kp * din_pad;
    if (i >= total) return;
    int k = (int)(i % din_pad);
    int c = (int)(i / din_pad);
    float v = (c < Kout && k < din) ? W[(long long)k * Kout + c] : 0.0f;
    Wt[i] = (__bf16)v;
}

// ---------------- WMMA GEMM: G[N,Kp] = A[N,K] x Wt^T (Wt is [Kp,K]) ----------
// Each wave computes a 16x32 output tile (one M-tile x two adjacent N-tiles):
// one A-fragment load feeds two independent WMMA accumulator chains, which
// halves A load instructions per WMMA and hides XDL latency. 8 waves/block
// share the same M-tile when NtB is a multiple of 8, so A fragments hit near cache across the block.
// K multiple of 32. Whole-wave guards only: EXEC is all-1s at every WMMA.
__global__ void __launch_bounds__(256)
k_gemm_wmma_bf16(const __bf16* __restrict__ A, const __bf16* __restrict__ Bt,
                 float* __restrict__ G, int Mtiles, int Ntiles, int K, int ldc) {
    int wid = threadIdx.x >> 5;
    int NtB = (Ntiles + 1) >> 1;                 // N-tile pairs per M row
    int t   = blockIdx.x * (blockDim.x >> 5) + wid;
    if (t >= Mtiles * NtB) return;               // whole-wave exit
    int tnb = t % NtB;
    int tm  = t / NtB;
    int tn0 = tnb * 2;
    const bool has2 = (tn0 + 1) < Ntiles;        // wave-uniform

    int lane = threadIdx.x & 31;
    int half = lane >> 4;
    int idx  = lane & 15;

    const __bf16* ap  = A  + (size_t)(tm * 16 + idx) * K + half * 16;
    const __bf16* bp0 = Bt + (size_t)(tn0 * 16 + idx) * K + half * 16;
    const __bf16* bp1 = bp0 + (size_t)16 * K;

    v8f acc0 = {}, acc1 = {};
    if (has2) {
        for (int k = 0; k < K; k += 32) {
            __builtin_prefetch((const void*)(ap  + k + 128), 0, 3);
            __builtin_prefetch((const void*)(bp0 + k + 128), 0, 3);
            __builtin_prefetch((const void*)(bp1 + k + 128), 0, 3);
            v16bf a  = *(const v16bf*)(ap + k);
            v16bf b0 = *(const v16bf*)(bp0 + k);
            v16bf b1 = *(const v16bf*)(bp1 + k);
            acc0 = __builtin_amdgcn_wmma_f32_16x16x32_bf16(
                       false, a, false, b0, (short)0, acc0, false, false);
            acc1 = __builtin_amdgcn_wmma_f32_16x16x32_bf16(
                       false, a, false, b1, (short)0, acc1, false, false);
        }
    } else {
        for (int k = 0; k < K; k += 32) {
            __builtin_prefetch((const void*)(ap  + k + 128), 0, 3);
            __builtin_prefetch((const void*)(bp0 + k + 128), 0, 3);
            v16bf a  = *(const v16bf*)(ap + k);
            v16bf b0 = *(const v16bf*)(bp0 + k);
            acc0 = __builtin_amdgcn_wmma_f32_16x16x32_bf16(
                       false, a, false, b0, (short)0, acc0, false, false);
        }
    }

    size_t rowbase = (size_t)(tm * 16 + half * 8) * ldc;
    int col0 = tn0 * 16 + idx;
#pragma unroll
    for (int r = 0; r < 8; ++r)
        G[rowbase + (size_t)r * ldc + col0] = acc0[r];
    if (has2) {
#pragma unroll
        for (int r = 0; r < 8; ++r)
            G[rowbase + (size_t)r * ldc + col0 + 16] = acc1[r];
    }
}

// ---------------- attention coefficients ----------------
// as[n,h] = sum_c G[n, h*C+c] * a_src[h*C+c]   (and same for ad)
__global__ void k_att_coef(const float* __restrict__ G,
                           const float* __restrict__ a_src,
                           const float* __restrict__ a_dst,
                           float* __restrict__ as_, float* __restrict__ ad_,
                           int N, int C, int ldc) {
    long long i = (long long)blockIdx.x * blockDim.x + threadIdx.x;
    if (i >= (long long)N * 2) return;
    int h = (int)(i & 1);
    long long n = i >> 1;
    const float* g = G + n * ldc + h * C;
    const float* vs = a_src + h * C;
    const float* vd = a_dst + h * C;
    float ss = 0.f, sd = 0.f;
    for (int c = 0; c < C; ++c) { float gv = g[c]; ss += gv * vs[c]; sd += gv * vd[c]; }
    as_[i] = ss;
    ad_[i] = sd;
}

// ---------------- edge kernels ----------------

__device__ __forceinline__ void atomicMaxF(float* addr, float v) {
    if (v >= 0.0f) atomicMax((int*)addr, __float_as_int(v));
    else           atomicMin((unsigned int*)addr, (unsigned int)__float_as_int(v));
}

// e = leaky_relu(as[src]+ad[dst]); store; atomic segment max into m[dst]
__global__ void k_edge_logits_max(const int* __restrict__ ei, int E, int N,
                                  const float* __restrict__ as_,
                                  const float* __restrict__ ad_,
                                  float* __restrict__ elog, float* __restrict__ m) {
    long long ET = (long long)E + N;
    long long i = (long long)blockIdx.x * blockDim.x + threadIdx.x;
    if (i >= ET * 2) return;
    int h = (int)(i & 1);
    long long e = i >> 1;
    int s, d;
    if (e < E) { s = ei[e]; d = ei[(long long)E + e]; }
    else       { s = d = (int)(e - E); }                 // self loops
    float v = as_[(long long)s * 2 + h] + ad_[(long long)d * 2 + h];
    v = (v > 0.f) ? v : NEG_SLOPE * v;
    elog[i] = v;
    atomicMaxF(&m[(long long)d * 2 + h], v);
}

// alpha_raw = exp(e - m[dst]); store; atomic segment sum into s[dst]
__global__ void k_edge_exp_sum(const int* __restrict__ ei, int E, int N,
                               const float* __restrict__ m,
                               float* __restrict__ elog, float* __restrict__ sb) {
    long long ET = (long long)E + N;
    long long i = (long long)blockIdx.x * blockDim.x + threadIdx.x;
    if (i >= ET * 2) return;
    int h = (int)(i & 1);
    long long e = i >> 1;
    int d = (e < E) ? ei[(long long)E + e] : (int)(e - E);
    float v = __expf(elog[i] - m[(long long)d * 2 + h]);
    elog[i] = v;
    atomicAdd(&sb[(long long)d * 2 + h], v);
}

// out[dst, h*C+c] += g[src, h*C+c] * alpha_raw/(sum+eps) ; one block per edge
__global__ void k_edge_scatter(const int* __restrict__ ei, int E, int N,
                               const float* __restrict__ elog,
                               const float* __restrict__ sb,
                               const float* __restrict__ G,
                               float* __restrict__ out, int C, int ldc) {
    long long e = blockIdx.x;
    int s, d;
    if (e < E) { s = ei[e]; d = ei[(long long)E + e]; }
    else       { s = d = (int)(e - E); }
    float w0 = elog[e * 2 + 0] / (sb[(long long)d * 2 + 0] + 1e-16f);
    float w1 = elog[e * 2 + 1] / (sb[(long long)d * 2 + 1] + 1e-16f);
    const float* gs = G + (long long)s * ldc;
    float* od = out + (long long)d * ldc;
    int Hd = 2 * C;
    for (int f = threadIdx.x; f < Hd; f += blockDim.x) {
        float w = (f < C) ? w0 : w1;
        atomicAdd(&od[f], gs[f] * w);
    }
}

// h_next[n,c] = 0.5*(out[n,c] + out[n,C+c]) + b[c]; optional ReLU
__global__ void k_finalize(const float* __restrict__ out, const float* __restrict__ b,
                           float* __restrict__ hn, int N, int C, int ldc, int do_relu) {
    long long i = (long long)blockIdx.x * blockDim.x + threadIdx.x;
    if (i >= (long long)N * C) return;
    int c = (int)(i % C);
    long long n = i / C;
    float v = 0.5f * (out[n * ldc + c] + out[n * ldc + C + c]) + b[c];
    if (do_relu) v = fmaxf(v, 0.f);
    hn[i] = v;
}

// row-wise softmax over 2 classes -> d_out
__global__ void k_softmax2(const float* __restrict__ h, float* __restrict__ o, int N) {
    long long n = (long long)blockIdx.x * blockDim.x + threadIdx.x;
    if (n >= N) return;
    float t0 = h[n * 2 + 0], t1 = h[n * 2 + 1];
    float mx = fmaxf(t0, t1);
    float e0 = __expf(t0 - mx), e1 = __expf(t1 - mx);
    float inv = 1.0f / (e0 + e1);
    o[n * 2 + 0] = e0 * inv;
    o[n * 2 + 1] = e1 * inv;
}

// ---------------- host-side orchestration ----------------

static inline int cdiv(long long a, long long b) { return (int)((a + b - 1) / b); }
static inline int rup(int a, int b) { return ((a + b - 1) / b) * b; }

extern "C" void kernel_launch(void* const* d_in, const int* in_sizes, int n_in,
                              void* d_out, int out_size, void* d_ws, size_t ws_size,
                              hipStream_t stream) {
    static const int DIN[8]  = {784, 4, 16, 32, 64, 128, 256, 512};
    static const int DOUT[8] = {4, 16, 32, 64, 128, 256, 512, 2};

    const float* x  = (const float*)d_in[0];
    const int*   ei = (const int*)d_in[1];
    const int N = in_sizes[0] / 784;   // 50000 (multiple of 16)
    const int E = in_sizes[1] / 2;     // 400000
    const long long ET = (long long)E + N;

    // workspace carve-out
    char* ws = (char*)d_ws;
    size_t off = 0;
    auto carve = [&](size_t bytes) -> void* {
        void* p = ws + off;
        off = (off + bytes + 255) & ~(size_t)255;
        return p;
    };
    __bf16* Abuf = (__bf16*)carve((size_t)N * 800 * sizeof(__bf16));
    __bf16* Wt   = (__bf16*)carve((size_t)1024 * 800 * sizeof(__bf16));
    float*  G    = (float*)carve((size_t)N * 1024 * sizeof(float));
    float*  OUT  = (float*)carve((size_t)N * 1024 * sizeof(float));
    float*  Hb   = (float*)carve((size_t)N * 512 * sizeof(float));
    float*  as_  = (float*)carve((size_t)N * 2 * sizeof(float));
    float*  ad_  = (float*)carve((size_t)N * 2 * sizeof(float));
    float*  mb   = (float*)carve((size_t)N * 2 * sizeof(float));
    float*  sb   = (float*)carve((size_t)N * 2 * sizeof(float));
    float*  elog = (float*)carve((size_t)ET * 2 * sizeof(float));
    (void)ws_size; (void)n_in; (void)out_size;

    const float* h = x;
    for (int i = 0; i < 8; ++i) {
        const int din = DIN[i], C = DOUT[i];
        const int Hd = 2 * C;
        const int din_pad = rup(din, 32);
        const int Kp = rup(Hd < 16 ? 16 : Hd, 16);
        const float* W     = (const float*)d_in[3 + 4 * i];
        const float* a_src = (const float*)d_in[4 + 4 * i];
        const float* a_dst = (const float*)d_in[5 + 4 * i];
        const float* bias  = (const float*)d_in[6 + 4 * i];

        // 1) pack operands to bf16 (zero-padded)
        long long na = (long long)N * din_pad;
        k_conv_h_bf16<<<cdiv(na, 256), 256, 0, stream>>>(h, Abuf, N, din, din_pad);
        long long nw = (long long)Kp * din_pad;
        k_conv_w_bf16t<<<cdiv(nw, 256), 256, 0, stream>>>(W, Wt, din, Hd, din_pad, Kp);

        // 2) WMMA GEMM: G[N,Kp] = A x Wt^T  (16x32 tile per wave)
        const int Mtiles = N / 16, Ntiles = Kp / 16;
        const int NtB = (Ntiles + 1) / 2;
        const long long tiles = (long long)Mtiles * NtB;
        k_gemm_wmma_bf16<<<cdiv(tiles, 8), 256, 0, stream>>>(Abuf, Wt, G,
                                                             Mtiles, Ntiles, din_pad, Kp);

        // 3) attention coefficients per node
        k_att_coef<<<cdiv((long long)N * 2, 256), 256, 0, stream>>>(G, a_src, a_dst,
                                                                    as_, ad_, N, C, Kp);

        // 4) segment softmax scratch init
        k_fill_f32<<<cdiv((long long)N * 2, 256), 256, 0, stream>>>(mb, -3.0e38f, (long long)N * 2);
        k_fill_f32<<<cdiv((long long)N * 2, 256), 256, 0, stream>>>(sb, 0.0f, (long long)N * 2);
        k_fill_f32<<<cdiv((long long)N * Kp, 256), 256, 0, stream>>>(OUT, 0.0f, (long long)N * Kp);

        // 5) edge phase: logits+max, exp+sum, weighted scatter
        k_edge_logits_max<<<cdiv(ET * 2, 256), 256, 0, stream>>>(ei, E, N, as_, ad_, elog, mb);
        k_edge_exp_sum<<<cdiv(ET * 2, 256), 256, 0, stream>>>(ei, E, N, mb, elog, sb);
        k_edge_scatter<<<(unsigned)ET, 128, 0, stream>>>(ei, E, N, elog, sb, G, OUT, C, Kp);

        // 6) head mean + bias (+ReLU on hidden layers)
        k_finalize<<<cdiv((long long)N * C, 256), 256, 0, stream>>>(OUT, bias, Hb, N, C, Kp,
                                                                    (i < 7) ? 1 : 0);
        h = Hb;
    }

    // final row softmax over 2 classes
    k_softmax2<<<cdiv((long long)N, 256), 256, 0, stream>>>(Hb, (float*)d_out, N);
}
